// HybridPhysicsModel_11982958756147
// MI455X (gfx1250) — matrix operand order repaired and checked by reading, not compiled, not run on hardware
//
#include <hip/hip_runtime.h>

#define PP  512
#define ZZ  4096
#define TT  16384
#define EE  65536
#define ZW  128           // ZZ/32 bitmask words per adjacency row
#define PEN 10.0f

typedef __attribute__((ext_vector_type(16))) _Float16 v16h;
typedef __attribute__((ext_vector_type(8)))  float    v8f;

// ---------------- adjacency bitmask ----------------
__global__ void k_zero(uint4* __restrict__ p, int n) {
    int i = blockIdx.x * blockDim.x + threadIdx.x;
    if (i < n) p[i] = uint4{0u, 0u, 0u, 0u};
}

__global__ void k_adj(const long long* __restrict__ ei, unsigned* __restrict__ adj) {
    int i = blockIdx.x * blockDim.x + threadIdx.x;
    if (i < EE) {
        int a = (int)ei[i];
        int b = (int)ei[EE + i];
        atomicOr(&adj[a * ZW + (b >> 5)], 1u << (b & 31));
        atomicOr(&adj[b * ZW + (a >> 5)], 1u << (a & 31));
    } else if (i < EE + ZZ) {
        int z = i - EE;
        atomicOr(&adj[z * ZW + (z >> 5)], 1u << (z & 31));
    }
}

// ---------------- layer-1 collapse: base = pa @ W1[:512] + b1 ----------------
__global__ void k_prep(const float* __restrict__ pa, const float* __restrict__ W1,
                       const float* __restrict__ b1,
                       float* __restrict__ base, float* __restrict__ w1last) {
    int j = threadIdx.x;
    if (j < 128) {
        float s = b1[j];
        for (int p = 0; p < PP; ++p) s += pa[p] * W1[p * 128 + j];
        base[j]   = s;
        w1last[j] = W1[PP * 128 + j];
    }
}

// ---------------- f32 -> f16 conversion into WMMA B-fragment order ----------------
// B-fragment order used here (32x16 f16, contiguous k): lane L holds N = L%16, element i holds
// K = (L>=16)*16 + i.  Each lane's 16 halves stored contiguously (32 B) so a
// fragment is one v16h load, contiguous 1 KB across the wave.
__global__ void k_cvt(const float* __restrict__ W2, const float* __restrict__ W3,
                      _Float16* __restrict__ W2f, _Float16* __restrict__ W3f) {
    int o = blockIdx.x * blockDim.x + threadIdx.x;
    const int n2 = 128 * 64;
    const int n3 = 64 * ZZ;
    if (o < n2) {
        int i    = o & 15;
        int lane = (o >> 4) & 31;
        int ks   = (o >> 9) & 3;      // 4 k-steps (K=128)
        int nt   = o >> 11;           // 4 n-tiles (N=64)
        int k = ks * 32 + ((lane >> 4) << 4) + i;
        int n = nt * 16 + (lane & 15);
        W2f[o] = (_Float16)W2[k * 64 + n];
    } else if (o < n2 + n3) {
        int o3 = o - n2;
        int i    = o3 & 15;
        int lane = (o3 >> 4) & 31;
        int ks   = (o3 >> 9) & 1;     // 2 k-steps (K=64)
        int nt   = o3 >> 10;          // 256 n-tiles (N=4096)
        int k = ks * 32 + ((lane >> 4) << 4) + i;
        int n = nt * 16 + (lane & 15);
        W3f[o3] = (_Float16)W3[k * ZZ + n];
    }
}

// ---------------- fused MLP: 16 timesteps per block, WMMA GEMM2+GEMM3 ----------------
__global__ void __launch_bounds__(256) k_mlp(
        const float* __restrict__ times,
        const float* __restrict__ base, const float* __restrict__ w1last,
        const _Float16* __restrict__ W2f, const float* __restrict__ b2,
        const _Float16* __restrict__ W3f, const float* __restrict__ b3,
        float* __restrict__ out) {
    __shared__ _Float16 h1s[16 * 128];
    __shared__ _Float16 h2s[16 * 64];
    const int tid = threadIdx.x;
    const int t0  = blockIdx.x * 16;

    // h1[row][j] = relu(base[j] + t * W1[512][j])
    for (int e = tid; e < 16 * 128; e += 256) {
        int row = e >> 7, j = e & 127;
        float v = base[j] + times[t0 + row] * w1last[j];
        h1s[e] = (_Float16)(v > 0.f ? v : 0.f);
    }
    __syncthreads();

    const int wave = tid >> 5;
    const int lane = tid & 31;
    const int m    = lane & 15;
    const int hi   = lane >> 4;
    const v16h* W2v = (const v16h*)W2f;
    const v16h* W3v = (const v16h*)W3f;

    // GEMM2: h2[16x64] = relu(h1[16x128] @ W2 + b2) ; waves 0..3 own one N-tile each
    if (wave < 4) {
        v8f c = {};
#pragma unroll
        for (int ks = 0; ks < 4; ++ks) {
            v16h a;
#pragma unroll
            for (int i = 0; i < 16; ++i) {
                int k = hi * 16 + i;
                a[i] = h1s[m * 128 + ks * 32 + k];
            }
            v16h b = W2v[(wave * 4 + ks) * 32 + lane];
            c = __builtin_amdgcn_wmma_f32_16x16x32_f16(false, a, false, b,
                                                       (short)0, c, false, false);
        }
        float bb = b2[wave * 16 + m];
#pragma unroll
        for (int r = 0; r < 8; ++r) {
            float v = c[r] + bb;
            h2s[(r + hi * 8) * 64 + wave * 16 + m] = (_Float16)(v > 0.f ? v : 0.f);
        }
    }
    __syncthreads();

    // GEMM3: raw[16x4096] = h2[16x64] @ W3 + b3 ; 8 waves x 32 N-tiles
    v16h a0, a1;
#pragma unroll
    for (int i = 0; i < 16; ++i) {
        int k = hi * 16 + i;
        a0[i] = h2s[m * 64 + k];
        a1[i] = h2s[m * 64 + 32 + k];
    }
    for (int jt = 0; jt < 32; ++jt) {
        const int nt = wave * 32 + jt;
        v16h bf0 = W3v[(nt * 2 + 0) * 32 + lane];
        v16h bf1 = W3v[(nt * 2 + 1) * 32 + lane];
        v8f c = {};
        c = __builtin_amdgcn_wmma_f32_16x16x32_f16(false, a0, false, bf0,
                                                   (short)0, c, false, false);
        c = __builtin_amdgcn_wmma_f32_16x16x32_f16(false, a1, false, bf1,
                                                   (short)0, c, false, false);
        float bb = b3[nt * 16 + m];
#pragma unroll
        for (int r = 0; r < 8; ++r)
            out[(size_t)(t0 + r + hi * 8) * ZZ + nt * 16 + m] = c[r] + bb;
    }
}

// Monotone (value, index) key: greater key == greater value, ties -> lower index.
__device__ __forceinline__ unsigned long long mkKey(float v, int z) {
    unsigned b = __float_as_uint(v);
    b = (b & 0x80000000u) ? ~b : (b | 0x80000000u);
    return ((unsigned long long)b << 32) | (unsigned)(ZZ - 1 - z);
}

// ---------------- sequential masked-argmax scan: 1 barrier + ds_max_u64 per step ----------------
__global__ void __launch_bounds__(1024) k_seq(const float* __restrict__ raw,
                                              const unsigned* __restrict__ adj,
                                              int* __restrict__ curArr) {
    __shared__ unsigned long long acc[3];   // rotating argmax accumulators
    const int tid  = threadIdx.x;
    const int lane = tid & 31;
    if (tid < 3) acc[tid] = 0ull;
    __syncthreads();

    float4 r4 = ((const float4*)raw)[tid];  // stage-0 preload
    int cur = 0;

    for (int t = 0; t < TT; ++t) {
        if (tid == 0) curArr[t] = cur;

        // next row's load + prefetch issued before any state-dependent work
        float4 nxt{0.f, 0.f, 0.f, 0.f};
        if (t + 1 < TT) nxt = ((const float4*)(raw + (size_t)(t + 1) * ZZ))[tid];
        if (t + 4 < TT) __builtin_prefetch(raw + (size_t)(t + 4) * ZZ + tid * 4, 0, 0);

        unsigned w = adj[cur * ZW + (tid >> 3)];   // 4 bits for this thread's 4 zones
        float vals[4] = {r4.x, r4.y, r4.z, r4.w};
        float bv = -1e30f; int bi = 0;
#pragma unroll
        for (int k = 0; k < 4; ++k) {
            float v = vals[k] - (((w >> (((tid & 7) << 2) + k)) & 1u) ? 0.f : PEN);
            if (v > bv) { bv = v; bi = tid * 4 + k; }
        }
#pragma unroll
        for (int off = 16; off > 0; off >>= 1) {
            float ov = __shfl_down(bv, off, 32);
            int   oi = __shfl_down(bi, off, 32);
            if (ov > bv || (ov == bv && oi < bi)) { bv = ov; bi = oi; }
        }
        if (lane == 0) atomicMax(&acc[t % 3], mkKey(bv, bi));
        // reset buffer for iter t+1: last read at iter t-2 (ordered by t-1's barrier),
        // next atomics at iter t+1 (ordered by this barrier) -> no extra barrier needed
        if (tid == 0 && t + 1 < TT) acc[(t + 1) % 3] = 0ull;
        __syncthreads();
        cur = ZZ - 1 - (int)(acc[t % 3] & 0xFFFFFFFFull);
        r4 = nxt;
    }
}

// ---------------- parallel in-place penalty apply: out -= 10*(1-adj[cur[t]]) ----------------
__global__ void __launch_bounds__(256) k_fix(float* __restrict__ out,
                                             const unsigned* __restrict__ adj,
                                             const int* __restrict__ curArr) {
    const int t  = blockIdx.y;
    const int z0 = (blockIdx.x * 256 + threadIdx.x) * 4;
    const int cur = curArr[t];
    unsigned w = adj[cur * ZW + (z0 >> 5)];
    float4* p = (float4*)(out + (size_t)t * ZZ + z0);
    float4 v = *p;
    int s = z0 & 31;
    v.x -= ((w >> (s + 0)) & 1u) ? 0.f : PEN;
    v.y -= ((w >> (s + 1)) & 1u) ? 0.f : PEN;
    v.z -= ((w >> (s + 2)) & 1u) ? 0.f : PEN;
    v.w -= ((w >> (s + 3)) & 1u) ? 0.f : PEN;
    *p = v;
}

extern "C" void kernel_launch(void* const* d_in, const int* in_sizes, int n_in,
                              void* d_out, int out_size, void* d_ws, size_t ws_size,
                              hipStream_t stream) {
    (void)in_sizes; (void)n_in; (void)out_size; (void)ws_size;
    const float*     pa    = (const float*)d_in[0];
    const float*     times = (const float*)d_in[1];
    /* d_in[2] zone_features is unused by the reference */
    const long long* ei    = (const long long*)d_in[3];   // int64 edge_index [2,E]
    const float*     W1    = (const float*)d_in[4];
    const float*     b1    = (const float*)d_in[5];
    const float*     W2    = (const float*)d_in[6];
    const float*     b2    = (const float*)d_in[7];
    const float*     W3    = (const float*)d_in[8];
    const float*     b3    = (const float*)d_in[9];
    float*           out   = (float*)d_out;

    // workspace layout (~2.7 MB)
    char*     ws     = (char*)d_ws;
    unsigned* adj    = (unsigned*)ws;                                   // 2 MB bitmask
    float*    base   = (float*)(ws + 2 * 1024 * 1024);                  // 128 f32
    float*    w1last = base + 128;                                      // 128 f32
    _Float16* W2f    = (_Float16*)(ws + 2 * 1024 * 1024 + 1024);        // 16 KB (frag order)
    _Float16* W3f    = W2f + 128 * 64;                                  // 512 KB (frag order)
    int*      curArr = (int*)(ws + 2 * 1024 * 1024 + 1024 +
                              (size_t)(128 * 64 + 64 * ZZ) * 2);        // 64 KB

    k_zero<<<512, 256, 0, stream>>>((uint4*)adj, (ZZ * ZW * 4) / 16);
    k_adj <<<(EE + ZZ + 255) / 256, 256, 0, stream>>>(ei, adj);
    k_prep<<<1, 128, 0, stream>>>(pa, W1, b1, base, w1last);
    k_cvt <<<(128 * 64 + 64 * ZZ + 255) / 256, 256, 0, stream>>>(W2, W3, W2f, W3f);
    k_mlp <<<TT / 16, 256, 0, stream>>>(times, base, w1last, W2f, b2, W3f, b3, out);
    k_seq <<<1, 1024, 0, stream>>>(out, adj, curArr);
    dim3 gfix(ZZ / 1024, TT);
    k_fix <<<gfix, 256, 0, stream>>>(out, adj, curArr);
}
